// APPNP1_16638703304886
// MI455X (gfx1250) — matrix operand order repaired and checked by reading, not compiled, not run on hardware
//
#include <hip/hip_runtime.h>
#include <hip/hip_bf16.h>

// ---------------------------------------------------------------------------
// APPNP on MI455X (gfx1250, wave32):
//   h1 = relu(x @ W1 + b1)   v_wmma_f32_16x16x32_f16, B staged via TDM->LDS
//   h  = h1 @ W2 + b2        v_wmma_f32_16x16x32_f16
//   z  = h; 10x: z = 0.9*A z + 0.1*h   edge-scatter global_atomic_add_f32
//   out = log_softmax(z)     wave32 shuffle reduction
// ---------------------------------------------------------------------------

#define NNODES 100000
#define NEDGES 1600000
#define NFEAT  512
#define NHID   256
#define NCLASS 40
#define KSTEPS 10
#define ALPHA  0.1f

typedef __attribute__((ext_vector_type(16))) _Float16 v16h;
typedef __attribute__((ext_vector_type(8)))  _Float16 v8h;
typedef __attribute__((ext_vector_type(8)))  float    v8f;
typedef __attribute__((ext_vector_type(4)))  unsigned int u32x4;
typedef __attribute__((ext_vector_type(8)))  int      i32x8;
typedef __attribute__((ext_vector_type(4)))  int      i32x4;

// ---------------------------------------------------------------------------
// Pre-swizzle W1 (f32 [512,256] row-major, K x N) into WMMA B-fragment order:
// W1p[kt][nt][lane][i] (halfs). B layout (16-bit, 32x16): lanes 0-15 hold
// column n=lane with K = kt*32 + {0..7,16..23}; lanes 16-31 hold K = kt*32 + {8..15,24..31}.
// ---------------------------------------------------------------------------
__global__ __launch_bounds__(256) void prep_w1(const float* __restrict__ W1,
                                               _Float16* __restrict__ W1p) {
    int t = blockIdx.x * 256 + threadIdx.x;            // 16*16*32*16 = 131072
    if (t >= 16 * 16 * 32 * 16) return;
    int i  = t & 15;
    int l  = (t >> 4) & 31;
    int nt = (t >> 9) & 15;
    int kt = t >> 13;
    int n  = nt * 16 + (l & 15);
    int k  = kt * 32 + (l >> 4) * 8 + (i & 7) + (i >> 3) * 16;
    W1p[t] = (_Float16)W1[(size_t)k * NHID + n];
}

// W2 (f32 [256,40]) -> W2p[kt(8)][nt(3)][lane(32)][i(16)], zero-padded n>=40
__global__ __launch_bounds__(256) void prep_w2(const float* __restrict__ W2,
                                               _Float16* __restrict__ W2p) {
    int t = blockIdx.x * 256 + threadIdx.x;            // 8*3*32*16 = 12288
    if (t >= 8 * 3 * 32 * 16) return;
    int i   = t & 15;
    int l   = (t >> 4) & 31;
    int rem = t >> 9;
    int nt  = rem % 3;
    int kt  = rem / 3;
    int n   = nt * 16 + (l & 15);
    int k   = kt * 32 + (l >> 4) * 8 + (i & 7) + (i >> 3) * 16;
    W2p[t] = (n < NCLASS) ? (_Float16)W2[(size_t)k * NCLASS + n] : (_Float16)0.0f;
}

// ---------------------------------------------------------------------------
// GEMM1: h1[N,256] (f16) = relu(x[N,512] @ W1 + b1)
// Block = 8 waves on one n-group (4 ntiles); per block, the needed W1p slice
// (16 kt-rows x 4KB, row stride 16KB = 64KB) is DMA'd into LDS once by the
// Tensor Data Mover, then the WMMA loop feeds B from LDS (ds_load).
// Grid: ceil(6250/8)=782 m-groups x 4 n-groups = 3128 blocks.
// ---------------------------------------------------------------------------
__global__ __launch_bounds__(256) void gemm1_wmma(const float* __restrict__ x,
                                                  const _Float16* __restrict__ W1p,
                                                  const float* __restrict__ b1,
                                                  _Float16* __restrict__ h1) {
    extern __shared__ _Float16 smem[];   // 64KB: [kt(16)][j(4)][lane(32)][i(16)]
    const int lane = threadIdx.x & 31;
    const int ng   = blockIdx.x & 3;                       // n-group (4 ntiles)
    const int mt   = (blockIdx.x >> 2) * 8 + (threadIdx.x >> 5);

    // --- TDM: stage this block's weight slice into LDS (wave 0 only) -------
    if (threadIdx.x < 32) {
        // D# group0: count=1, lds_addr=0 (dynamic LDS base), global byte addr,
        // type=2 ("image") in bits 127:126.
        unsigned long long ga =
            (unsigned long long)(uintptr_t)W1p + (unsigned)(ng * 4 * 512 * 2);
        u32x4 g0;
        g0[0] = 1u;                                        // count=1
        g0[1] = 0u;                                        // lds_addr
        g0[2] = (unsigned)ga;                              // global_addr[31:0]
        g0[3] = (unsigned)((ga >> 32) & 0x01FFFFFFu) | (2u << 30);
        // D# group1 (8B data units): tile 512x16, tensor dim0=2048 stride=2048
        i32x8 g1;
        g1[0] = (3 << 16);                                 // data_size = 8B
        g1[1] = (int)(2048u << 16);                        // tensor_dim0[15:0]
        g1[2] = (16 << 16);                                // tensor_dim1[15:0]
        g1[3] = (int)(512u << 16);                         // tile_dim0 = 512
        g1[4] = 16;                                        // tile_dim1 = 16
        g1[5] = 2048;                                      // dim0_stride[31:0]
        g1[6] = 0;
        g1[7] = 0;
        i32x4 gz4 = {0, 0, 0, 0};                          // 2D: groups 2/3 off
        i32x8 gz8 = {0, 0, 0, 0, 0, 0, 0, 0};
        __builtin_amdgcn_tensor_load_to_lds(g0, g1, gz4, gz4, gz8, 0);
        __builtin_amdgcn_s_wait_tensorcnt(0);
    }
    __syncthreads();
    if (mt >= NNODES / 16) return;                         // wave-uniform

    const int sel  = lane >> 4;
    const int mrow = mt * 16 + (lane & 15);
    const float* arow = x + (size_t)mrow * NFEAT + sel * 8;
    const v16h*  bls  = (const v16h*)smem;

    v8f acc0 = {}, acc1 = {}, acc2 = {}, acc3 = {};

    for (int kt = 0; kt < NFEAT / 32; ++kt) {
        // A fragment: ISA 16-bit A 16x32 layout, f32->f16 convert on the fly.
        const float* p0 = arow + kt * 32;
        const float* p1 = p0 + 16;
        v16h a;
#pragma unroll
        for (int i = 0; i < 8; ++i) {
            a[i]     = (_Float16)p0[i];
            a[8 + i] = (_Float16)p1[i];
        }
        const v16h* bp = bls + (size_t)(kt * 4) * 32 + lane;
        v16h b0  = bp[0];
        v16h b1v = bp[32];
        v16h b2v = bp[64];
        v16h b3v = bp[96];
        acc0 = __builtin_amdgcn_wmma_f32_16x16x32_f16(false, a, false, b0,  (short)0, acc0, false, false);
        acc1 = __builtin_amdgcn_wmma_f32_16x16x32_f16(false, a, false, b1v, (short)0, acc1, false, false);
        acc2 = __builtin_amdgcn_wmma_f32_16x16x32_f16(false, a, false, b2v, (short)0, acc2, false, false);
        acc3 = __builtin_amdgcn_wmma_f32_16x16x32_f16(false, a, false, b3v, (short)0, acc3, false, false);
    }

    // C/D layout: acc[r] holds (row = mt*16 + r + 8*sel, col = lane&15)
    const float bias = b1[0];
    v8f accs[4] = {acc0, acc1, acc2, acc3};
#pragma unroll
    for (int j = 0; j < 4; ++j) {
        int n = (ng * 4 + j) * 16 + (lane & 15);
#pragma unroll
        for (int r = 0; r < 8; ++r) {
            int rowm = mt * 16 + r + sel * 8;
            float v = accs[j][r] + bias;
            h1[(size_t)rowm * NHID + n] = (_Float16)(v > 0.0f ? v : 0.0f);
        }
    }
}

// ---------------------------------------------------------------------------
// GEMM2: h[N,40] (f32) = h1[N,256] @ W2 + b2   (N padded to 48 = 3 ntiles)
// ---------------------------------------------------------------------------
__global__ __launch_bounds__(256) void gemm2_wmma(const _Float16* __restrict__ h1,
                                                  const v16h* __restrict__ W2p,
                                                  const float* __restrict__ b2,
                                                  float* __restrict__ h) {
    const int lane = threadIdx.x & 31;
    const int mt   = blockIdx.x * 8 + (threadIdx.x >> 5);
    if (mt >= NNODES / 16) return;                       // wave-uniform branch
    const int sel  = lane >> 4;
    const int mrow = mt * 16 + (lane & 15);

    const _Float16* arow = h1 + (size_t)mrow * NHID + sel * 8;

    v8f acc0 = {}, acc1 = {}, acc2 = {};

    for (int kt = 0; kt < NHID / 32; ++kt) {
        v8h lo = *(const v8h*)(arow + kt * 32);
        v8h hi = *(const v8h*)(arow + kt * 32 + 16);
        v16h a;
#pragma unroll
        for (int i = 0; i < 8; ++i) { a[i] = lo[i]; a[8 + i] = hi[i]; }
        const v16h* bp = W2p + (size_t)(kt * 3) * 32 + lane;
        v16h b0  = bp[0];
        v16h b1v = bp[32];
        v16h b2v = bp[64];
        acc0 = __builtin_amdgcn_wmma_f32_16x16x32_f16(false, a, false, b0,  (short)0, acc0, false, false);
        acc1 = __builtin_amdgcn_wmma_f32_16x16x32_f16(false, a, false, b1v, (short)0, acc1, false, false);
        acc2 = __builtin_amdgcn_wmma_f32_16x16x32_f16(false, a, false, b2v, (short)0, acc2, false, false);
    }

    const float bias = b2[0];
    v8f accs[3] = {acc0, acc1, acc2};
#pragma unroll
    for (int j = 0; j < 3; ++j) {
        int n = j * 16 + (lane & 15);
        if (n < NCLASS) {
#pragma unroll
            for (int r = 0; r < 8; ++r) {
                int rowm = mt * 16 + r + sel * 8;
                h[(size_t)rowm * NCLASS + n] = accs[j][r] + bias;
            }
        }
    }
}

// z_out[i] = alpha * h[i]
__global__ __launch_bounds__(256) void init_z(const float* __restrict__ h,
                                              float* __restrict__ zout, int n) {
    int i = blockIdx.x * 256 + threadIdx.x;
    if (i < n) zout[i] = ALPHA * h[i];
}

// Edge-scatter SpMM: zout[row] += (1-alpha)*vals * zin[col]; one edge / wave32.
__global__ __launch_bounds__(256) void spmm_scatter(const int* __restrict__ row,
                                                    const int* __restrict__ col,
                                                    const float* __restrict__ vals,
                                                    const float* __restrict__ zin,
                                                    float* __restrict__ zout,
                                                    int nedges) {
    const int lane = threadIdx.x & 31;
    const int e    = blockIdx.x * 8 + (threadIdx.x >> 5);
    if (e >= nedges) return;
    const int r   = row[e];
    const int c   = col[e];
    const float v = vals[e] * (1.0f - ALPHA);
    const float* zc = zin  + (size_t)c * NCLASS;
    float*       zr = zout + (size_t)r * NCLASS;
    unsafeAtomicAdd(&zr[lane], v * zc[lane]);
    if (lane < NCLASS - 32)
        unsafeAtomicAdd(&zr[32 + lane], v * zc[32 + lane]);
}

// log_softmax over 40 classes: one wave32 per node, shfl_xor reductions.
__global__ __launch_bounds__(256) void log_softmax_rows(const float* __restrict__ z,
                                                        float* __restrict__ out) {
    const int lane = threadIdx.x & 31;
    const int node = blockIdx.x * 8 + (threadIdx.x >> 5);
    if (node >= NNODES) return;
    const float* zr = z + (size_t)node * NCLASS;
    float v0 = zr[lane];
    float v1 = (lane < NCLASS - 32) ? zr[32 + lane] : -3.402823e38f;
    float m  = fmaxf(v0, v1);
#pragma unroll
    for (int off = 16; off > 0; off >>= 1)
        m = fmaxf(m, __shfl_xor(m, off, 32));
    float s = expf(v0 - m) + ((lane < NCLASS - 32) ? expf(v1 - m) : 0.0f);
#pragma unroll
    for (int off = 16; off > 0; off >>= 1)
        s += __shfl_xor(s, off, 32);
    float ls = logf(s);
    float* o = out + (size_t)node * NCLASS;
    o[lane] = v0 - m - ls;
    if (lane < NCLASS - 32) o[32 + lane] = v1 - m - ls;
}

// ---------------------------------------------------------------------------
extern "C" void kernel_launch(void* const* d_in, const int* in_sizes, int n_in,
                              void* d_out, int out_size, void* d_ws, size_t ws_size,
                              hipStream_t stream) {
    const float* x    = (const float*)d_in[0];
    const int*   row  = (const int*)  d_in[1];
    const int*   col  = (const int*)  d_in[2];
    const float* vals = (const float*)d_in[3];
    const float* W1   = (const float*)d_in[4];
    const float* b1   = (const float*)d_in[5];
    const float* W2   = (const float*)d_in[6];
    const float* b2   = (const float*)d_in[7];
    float*       out  = (float*)d_out;
    const int E = in_sizes[1];

    // Workspace carve-up (~95 MB total)
    char* ws = (char*)d_ws;
    _Float16* h1  = (_Float16*)ws;                                  // N*256 f16
    _Float16* W1p = h1  + (size_t)NNODES * NHID;                    // 131072 halfs
    _Float16* W2p = W1p + 16 * 16 * 32 * 16;                        // 12288 halfs
    float*    h   = (float*)(W2p + 8 * 3 * 32 * 16);                // N*40 f32
    float*    z0  = h  + (size_t)NNODES * NCLASS;
    float*    z1  = z0 + (size_t)NNODES * NCLASS;

    // 1) weight pre-swizzle into WMMA B-fragment order
    prep_w1<<<(16 * 16 * 32 * 16) / 256, 256, 0, stream>>>(W1, W1p);
    prep_w2<<<(8 * 3 * 32 * 16 + 255) / 256, 256, 0, stream>>>(W2, W2p);

    // 2) MLP via WMMA (GEMM1 stages weights with the Tensor Data Mover)
    const int mgroups = (NNODES / 16 + 7) / 8;            // 782
    gemm1_wmma<<<mgroups * 4, 256, 65536, stream>>>(x, W1p, b1, h1);
    gemm2_wmma<<<mgroups, 256, 0, stream>>>(h1, (const v16h*)W2p, b2, h);

    // 3) APPNP propagation: z <- (1-a)*A z + a*h, 10 steps, ping-pong in ws
    const int NZ = NNODES * NCLASS;
    const float* zin = h;            // scan carry starts at h
    float* za = z0;
    float* zb = z1;
    for (int s = 0; s < KSTEPS; ++s) {
        init_z<<<(NZ + 255) / 256, 256, 0, stream>>>(h, za, NZ);
        spmm_scatter<<<(E + 7) / 8, 256, 0, stream>>>(row, col, vals, zin, za, E);
        zin = za;
        float* t = za; za = zb; zb = t;
    }

    // 4) log_softmax
    log_softmax_rows<<<(NNODES + 7) / 8, 256, 0, stream>>>(zin, out);
}
